// AUXCPC_GRUDTXT_DIR_65420941853088
// MI455X (gfx1250) — matrix operand order repaired and checked by reading, not compiled, not run on hardware
//
#include <hip/hip_runtime.h>

// ---------------- model dims ----------------
#define Hd   256
#define NLr  4
#define NHh  8
#define DHd  32
#define FFd  512
#define Fd   16
#define T24d 24
#define TSd  12
#define Bd   256
#define Ld   128
#define CDIMd 514
#define CPAD  544        // CDIM padded to mult of 32
#define BLr  (Bd*Ld)     // 32768 rows

typedef __attribute__((ext_vector_type(16))) __bf16 v16bf;
typedef __attribute__((ext_vector_type(8)))  __bf16 v8bf;
typedef __attribute__((ext_vector_type(8)))  float  v8f;

enum { ACT_NONE = 0, ACT_RELU = 1, ACT_EXPNEGRELU = 2 };

static __device__ __forceinline__ v8f wmma_bf16(v16bf a, v16bf b, v8f c) {
    return __builtin_amdgcn_wmma_f32_16x16x32_bf16(false, a, false, b, (short)0, c,
                                                   false, false);
}

// ------------- weight prep: fp32 [K,N] -> bf16 [N,Kp] (zero-padded K) -------------
__global__ void transpose_bf16_kernel(const float* __restrict__ src, __bf16* __restrict__ dst,
                                      int K, int N, int Kp) {
    long idx = (long)blockIdx.x * blockDim.x + threadIdx.x;
    long total = (long)N * Kp;
    if (idx >= total) return;
    int n = (int)(idx / Kp);
    int k = (int)(idx - (long)n * Kp);
    dst[idx] = (k < K) ? (__bf16)src[(long)k * N + n] : (__bf16)0.f;
}

// fp32 [T,C,D] -> bf16 [(t*D+d)][Kp] with k = c (zero-padded)
__global__ void transpose_tcd_bf16_kernel(const float* __restrict__ src, __bf16* __restrict__ dst,
                                          int T, int C, int D, int Kp) {
    long idx = (long)blockIdx.x * blockDim.x + threadIdx.x;
    long total = (long)T * D * Kp;
    if (idx >= total) return;
    int n = (int)(idx / Kp);            // n = t*D + d
    int c = (int)(idx - (long)n * Kp);
    int t = n / D;
    int d = n - t * D;
    dst[idx] = (c < C) ? (__bf16)src[((long)t * C + c) * D + d] : (__bf16)0.f;
}

// ------------- generic bf16 WMMA GEMM -------------
// out = act(A[M,K] * Wt[N,Kp]^T + bias); each wave computes a 16x64 strip:
// one A fragment per K-chunk feeds 4 WMMAs (4x arithmetic-intensity vs 16x16).
__global__ void gemm_bf16_kernel(const float* __restrict__ A, int lda,
                                 const __bf16* __restrict__ Bt, int Kp,
                                 const float* __restrict__ bias,
                                 float* __restrict__ out, int ldo,
                                 int M, int N, int act) {
    int lane = threadIdx.x & 31;
    int wave = threadIdx.x >> 5;
    int ngN = N >> 6;                          // 64-wide column groups
    int tile = blockIdx.x * 8 + wave;
    int total = (M >> 4) * ngN;
    if (tile >= total) return;                 // whole-wave uniform exit
    int mt = tile / ngN;
    int ng = tile - mt * ngN;
    int row  = (mt << 4) + (lane & 15);
    int col0 = (ng << 6) + (lane & 15);
    int khalf = (lane >> 4) << 3;              // 0 or 8
    int kb    = (lane >> 4) << 4;              // 0 or 16
    const float*  ap  = A  + (long)row * lda;
    const __bf16* bp0 = Bt + (long)(col0     ) * Kp;
    const __bf16* bp1 = Bt + (long)(col0 + 16) * Kp;
    const __bf16* bp2 = Bt + (long)(col0 + 32) * Kp;
    const __bf16* bp3 = Bt + (long)(col0 + 48) * Kp;
    v8f acc0 = {}, acc1 = {}, acc2 = {}, acc3 = {};
    for (int kc = 0; kc < Kp; kc += 32) {
        const float* a0 = ap + kc + kb;
        __builtin_prefetch(a0 + 32, 0, 1);
        v16bf a;
#pragma unroll
        for (int j = 0; j < 8; ++j) a[j] = (__bf16)a0[j];
#pragma unroll
        for (int j = 0; j < 8; ++j) a[8 + j] = (__bf16)a0[8 + j];
        v16bf b0 = *(const v16bf*)(bp0 + kc + kb);
        v16bf b1 = *(const v16bf*)(bp1 + kc + kb);
        v16bf b2 = *(const v16bf*)(bp2 + kc + kb);
        v16bf b3 = *(const v16bf*)(bp3 + kc + kb);
        acc0 = wmma_bf16(a, b0, acc0);
        acc1 = wmma_bf16(a, b1, acc1);
        acc2 = wmma_bf16(a, b2, acc2);
        acc3 = wmma_bf16(a, b3, acc3);
    }
    int mrow = (mt << 4) + ((lane >> 4) << 3);
    v8f accs[4] = {acc0, acc1, acc2, acc3};
#pragma unroll
    for (int g = 0; g < 4; ++g) {
        int col = col0 + (g << 4);
        float bvv = bias ? bias[col] : 0.f;
#pragma unroll
        for (int i = 0; i < 8; ++i) {
            float v = accs[g][i] + bvv;
            if (act == ACT_RELU)            v = fmaxf(v, 0.f);
            else if (act == ACT_EXPNEGRELU) v = __expf(-fmaxf(v, 0.f));
            out[(long)(mrow + i) * ldo + col] = v;
        }
    }
}

// ------------- embedding gather -------------
__global__ void gather_emb_kernel(const int* __restrict__ txts, const float* __restrict__ emb,
                                  float* __restrict__ tx) {
    int idx = blockIdx.x * blockDim.x + threadIdx.x;   // BL * H
    if (idx >= BLr * Hd) return;
    int row = idx >> 8;
    int j = idx & 255;
    tx[idx] = emb[(long)txts[row] * Hd + j];
}

// ------------- LayerNorm over 256, one wave per row -------------
// mode 0: out = LN(x); mode 1: out = LN(x + r); mode 2: out = LN(x) + PE(row%L, d)
__global__ void ln256_kernel(const float* __restrict__ x, const float* __restrict__ r,
                             const float* __restrict__ g, const float* __restrict__ bb,
                             float* __restrict__ out, int nrows, int mode) {
    int lane = threadIdx.x & 31;
    int wave = threadIdx.x >> 5;
    int row = blockIdx.x * 8 + wave;
    if (row >= nrows) return;
    const float* xp = x + (long)row * 256;
    float v[8];
    float s = 0.f;
#pragma unroll
    for (int j = 0; j < 8; ++j) {
        float t = xp[j * 32 + lane];
        if (mode == 1) t += r[(long)row * 256 + j * 32 + lane];
        v[j] = t; s += t;
    }
#pragma unroll
    for (int off = 16; off > 0; off >>= 1) s += __shfl_xor(s, off, 32);
    float mu = s * (1.f / 256.f);
    float q = 0.f;
#pragma unroll
    for (int j = 0; j < 8; ++j) { float d = v[j] - mu; q += d * d; }
#pragma unroll
    for (int off = 16; off > 0; off >>= 1) q += __shfl_xor(q, off, 32);
    float inv = rsqrtf(q * (1.f / 256.f) + 1e-5f);
#pragma unroll
    for (int j = 0; j < 8; ++j) {
        int d = j * 32 + lane;
        float o = (v[j] - mu) * inv * g[d] + bb[d];
        if (mode == 2) {
            int l = row & (Ld - 1);
            int i2 = d & ~1;
            float ang = (float)l * __expf(-(float)i2 * (9.210340371976184f / 256.f));
            o += (d & 1) ? cosf(ang) : sinf(ang);
        }
        out[(long)row * 256 + d] = o;
    }
}

// ------------- attention: one block per (b, head), 8 waves -------------
__global__ void attn_kernel(const float* __restrict__ q, const float* __restrict__ k,
                            const float* __restrict__ v, const int* __restrict__ tlen,
                            float* __restrict__ o) {
    __shared__ __bf16 vt[DHd * Ld];        // V^T  [dh][l]
    __shared__ __bf16 pl[8 * 16 * Ld];     // P per wave [16][128]
    int bh = blockIdx.x;
    int b = bh / NHh;
    int h = bh - b * NHh;
    int lane = threadIdx.x & 31;
    int wave = threadIdx.x >> 5;

    const float* vb = v + ((long)b * Ld) * Hd + h * DHd;
    for (int idx = threadIdx.x; idx < DHd * Ld; idx += blockDim.x) {
        int dh = idx >> 7;
        int l  = idx & (Ld - 1);
        vt[idx] = (__bf16)vb[(long)l * Hd + dh];
    }
    __syncthreads();

    int r0 = wave << 4;
    int khalf = (lane >> 4) << 3;
    int kb    = (lane >> 4) << 4;
    int rowa = r0 + (lane & 15);
    const float* qrow = q + ((long)(b * Ld) + rowa) * Hd + h * DHd;
    v16bf a;
#pragma unroll
    for (int j = 0; j < 8; ++j) a[j]     = (__bf16)qrow[khalf + j];
#pragma unroll
    for (int j = 0; j < 8; ++j) a[8 + j] = (__bf16)qrow[16 + khalf + j];

    int len = tlen[b] + 2;
    const float scale = 0.17677669529663687f;   // 1/sqrt(32)
    float s[8][8];
#pragma unroll
    for (int j = 0; j < 8; ++j) {
        int col = (j << 4) + (lane & 15);
        const float* krow = k + ((long)(b * Ld) + col) * Hd + h * DHd;
        v16bf bfr;
#pragma unroll
        for (int t = 0; t < 16; ++t) bfr[t] = (__bf16)krow[(t < 8) ? (khalf + t) : (8 + khalf + t)];
        v8f cc = {};
        cc = wmma_bf16(a, bfr, cc);
        float add = (col < len) ? 0.f : -1e9f;
#pragma unroll
        for (int i = 0; i < 8; ++i) s[j][i] = cc[i] * scale + add;
    }
    // softmax: per element i, rows i (lanes 0-15) and i+8 (lanes 16-31)
#pragma unroll
    for (int i = 0; i < 8; ++i) {
        float m = s[0][i];
#pragma unroll
        for (int j = 1; j < 8; ++j) m = fmaxf(m, s[j][i]);
#pragma unroll
        for (int off = 1; off < 16; off <<= 1) m = fmaxf(m, __shfl_xor(m, off, 16));
        float sum = 0.f;
#pragma unroll
        for (int j = 0; j < 8; ++j) { s[j][i] = __expf(s[j][i] - m); sum += s[j][i]; }
#pragma unroll
        for (int off = 1; off < 16; off <<= 1) sum += __shfl_xor(sum, off, 16);
        float inv = 1.f / sum;
#pragma unroll
        for (int j = 0; j < 8; ++j) s[j][i] *= inv;
    }
    // stage P -> LDS bf16
    __bf16* pw = pl + wave * (16 * Ld);
    int rl = (lane >> 4) << 3;
#pragma unroll
    for (int j = 0; j < 8; ++j) {
        int col = (j << 4) + (lane & 15);
#pragma unroll
        for (int i = 0; i < 8; ++i) pw[(rl + i) * Ld + col] = (__bf16)s[j][i];
    }
    __syncthreads();

    // O = P @ V
    const __bf16* pr = pw + (lane & 15) * Ld;
    float* ob = o + ((long)(b * Ld)) * Hd + h * DHd;
#pragma unroll
    for (int nt = 0; nt < 2; ++nt) {
        v8f acc = {};
#pragma unroll
        for (int kc = 0; kc < Ld; kc += 32) {
            v8bf lo = *(const v8bf*)(pr + kc + kb);
            v8bf hi = *(const v8bf*)(pr + kc + kb + 8);
            v16bf pa = __builtin_shufflevector(lo, hi, 0, 1, 2, 3, 4, 5, 6, 7,
                                               8, 9, 10, 11, 12, 13, 14, 15);
            v16bf vb2 = *(const v16bf*)(vt + ((nt << 4) + (lane & 15)) * Ld + kc + kb);
            acc = wmma_bf16(pa, vb2, acc);
        }
        int mrow = r0 + rl;
#pragma unroll
        for (int i = 0; i < 8; ++i)
            ob[(long)(mrow + i) * Hd + (nt << 4) + (lane & 15)] = acc[i];
    }
}

// ------------- GRU-D preprocessing: build U [T*B, 32] and Dbuf [T*B, 32] -------------
__global__ void xprep_kernel(const float* __restrict__ x1, const float* __restrict__ m1,
                             const float* __restrict__ d1, const float* __restrict__ xm,
                             const float* __restrict__ decw, const float* __restrict__ decb,
                             float* __restrict__ U, float* __restrict__ Dbuf) {
    int idx = blockIdx.x * blockDim.x + threadIdx.x;    // T24*B*F
    if (idx >= T24d * Bd * Fd) return;
    int f = idx & (Fd - 1);
    int rem = idx >> 4;
    int b = rem & (Bd - 1);
    int t = rem >> 8;
    long src = ((long)b * 36 + t) * Fd + f;
    float xv = x1[src], mv = m1[src], dv = d1[src];
    float xd = __expf(-fmaxf(dv * decw[f] + decb[f], 0.f));
    float xn = mv * xv + (1.f - mv) * xd * xv + (1.f - mv) * (1.f - xd) * xm[f];
    long row = (long)t * Bd + b;
    U[row * 32 + f] = xn;
    U[row * 32 + 16 + f] = mv;
    Dbuf[row * 32 + f] = dv;
    Dbuf[row * 32 + 16 + f] = 0.f;
}

__global__ void copyf_kernel(const float* __restrict__ src, float* __restrict__ dst, int n) {
    int i = blockIdx.x * blockDim.x + threadIdx.x;
    if (i < n) dst[i] = src[i];
}

__global__ void gru_decay_kernel(const float* __restrict__ hdcy, const float* __restrict__ h,
                                 float* __restrict__ h1) {
    int i = blockIdx.x * blockDim.x + threadIdx.x;      // B*H
    if (i < Bd * Hd) h1[i] = hdcy[i] * h[i];
}

__global__ void gru_gate_kernel(const float* __restrict__ gx, const float* __restrict__ gh,
                                const float* __restrict__ h1, float* __restrict__ h,
                                float* __restrict__ yst) {
    int idx = blockIdx.x * blockDim.x + threadIdx.x;    // B*H
    if (idx >= Bd * Hd) return;
    int b = idx >> 8;
    int j = idx & 255;
    const float* gxr = gx + (long)b * 768;
    const float* ghr = gh + (long)b * 768;
    float r = 1.f / (1.f + __expf(-(gxr[j] + ghr[j])));
    float z = 1.f / (1.f + __expf(-(gxr[256 + j] + ghr[256 + j])));
    float n = tanhf(gxr[512 + j] + r * ghr[512 + j]);
    float hn = (1.f - z) * n + z * h1[idx];
    h[idx] = hn;
    yst[idx] = hn;
}

// ------------- c_t build + LN(514) -> zero-padded [B,544] -------------
__global__ void build_ct_kernel(const float* __restrict__ ys, const float* __restrict__ hdd,
                                const float* __restrict__ age, const float* __restrict__ gen,
                                const int* __restrict__ ilen,
                                const float* __restrict__ g, const float* __restrict__ bb,
                                float* __restrict__ cpad) {
    __shared__ float sred[256];
    int b = blockIdx.x;
    int tid = threadIdx.x;
    int t0 = ilen[b] - 1;
    float v[3];
#pragma unroll
    for (int p = 0; p < 3; ++p) {
        int idx = tid + p * 256;
        float val = 0.f;
        if (idx < 256)       val = ys[((long)t0 * Bd + b) * Hd + idx];
        else if (idx < 512)  val = hdd[((long)b * Ld) * Hd + (idx - 256)];
        else if (idx == 512) val = age[b];
        else if (idx == 513) val = gen[b];
        v[p] = val;
    }
    float s = v[0] + v[1] + v[2];
    sred[tid] = s; __syncthreads();
    for (int st = 128; st > 0; st >>= 1) { if (tid < st) sred[tid] += sred[tid + st]; __syncthreads(); }
    float mu = sred[0] / 514.f;
    __syncthreads();
    float q = 0.f;
#pragma unroll
    for (int p = 0; p < 3; ++p) {
        int idx = tid + p * 256;
        if (idx < CDIMd) { float d = v[p] - mu; q += d * d; }
    }
    sred[tid] = q; __syncthreads();
    for (int st = 128; st > 0; st >>= 1) { if (tid < st) sred[tid] += sred[tid + st]; __syncthreads(); }
    float inv = rsqrtf(sred[0] / 514.f + 1e-5f);
#pragma unroll
    for (int p = 0; p < 3; ++p) {
        int idx = tid + p * 256;
        if (idx < CPAD) {
            float o = (idx < CDIMd) ? (v[p] - mu) * inv * g[idx] + bb[idx] : 0.f;
            cpad[(long)b * CPAD + idx] = o;
        }
    }
}

// ------------- weighted MSE loss -------------
__global__ void loss_kernel(const float* __restrict__ pred, const float* __restrict__ x1,
                            const float* __restrict__ find, float* __restrict__ out_loss) {
    __shared__ float s1[256], s2[256];
    int tid = threadIdx.x;
    float ws = 0.f, wt = 0.f;
    for (int p = tid; p < Bd * TSd; p += 256) {
        int b = p / TSd;
        int t = p - b * TSd;
        float w = find[(long)b * TSd + t];
        float acc = 0.f;
#pragma unroll
        for (int f = 0; f < Fd; ++f) {
            float d = pred[(long)b * (TSd * Fd) + t * Fd + f] -
                      x1[((long)b * 36 + T24d + t) * Fd + f];
            acc += d * d;
        }
        ws += w * (acc * (1.f / Fd));
        wt += w;
    }
    s1[tid] = ws; s2[tid] = wt; __syncthreads();
    for (int st = 128; st > 0; st >>= 1) {
        if (tid < st) { s1[tid] += s1[tid + st]; s2[tid] += s2[tid + st]; }
        __syncthreads();
    }
    if (tid == 0) out_loss[0] = s1[0] / s2[0];
}

// ------------- batchnorm stats over batch -------------
__global__ void bnstat_kernel(const float* __restrict__ z1, float* __restrict__ mu,
                              float* __restrict__ var) {
    int idx = blockIdx.x * blockDim.x + threadIdx.x;    // TS*64 = 768
    if (idx >= TSd * 64) return;
    float s = 0.f, s2 = 0.f;
    for (int b = 0; b < Bd; ++b) {
        float t = z1[(long)b * 768 + idx];
        s += t; s2 += t * t;
    }
    float m = s * (1.f / Bd);
    mu[idx] = m;
    var[idx] = s2 * (1.f / Bd) - m * m;
}

// ------------- final head: relu(BN(z1)) @ fc_W2 + fc_b2 -------------
__global__ void head_kernel(const float* __restrict__ z1, const float* __restrict__ mu,
                            const float* __restrict__ var, const float* __restrict__ bng,
                            const float* __restrict__ bnb, const float* __restrict__ w2,
                            const float* __restrict__ b2, float* __restrict__ out) {
    int idx = blockIdx.x * blockDim.x + threadIdx.x;    // TS*B
    if (idx >= TSd * Bd) return;
    int t = idx >> 8;
    int b = idx & 255;
    float acc = 0.f;
#pragma unroll 4
    for (int d = 0; d < 64; ++d) {
        int c = t * 64 + d;
        float zn = (z1[(long)b * 768 + c] - mu[c]) * rsqrtf(var[c] + 1e-5f) * bng[c] + bnb[c];
        acc += fmaxf(zn, 0.f) * w2[c];
    }
    out[(long)t * Bd + b] = acc + b2[t];
}

// =======================================================================
static void gemm(hipStream_t s, const float* A, int lda, const __bf16* Bt, int Kp,
                 const float* bias, float* out, int ldo, int M, int N, int act) {
    int tiles = (M / 16) * (N / 64);
    int blocks = (tiles + 7) / 8;
    gemm_bf16_kernel<<<blocks, 256, 0, s>>>(A, lda, Bt, Kp, bias, out, ldo, M, N, act);
}

static void transpose_w(hipStream_t s, const float* src, __bf16* dst, int K, int N, int Kp) {
    long total = (long)N * Kp;
    int blocks = (int)((total + 255) / 256);
    transpose_bf16_kernel<<<blocks, 256, 0, s>>>(src, dst, K, N, Kp);
}

extern "C" void kernel_launch(void* const* d_in, const int* in_sizes, int n_in,
                              void* d_out, int out_size, void* d_ws, size_t ws_size,
                              hipStream_t stream) {
    (void)in_sizes; (void)n_in; (void)out_size; (void)ws_size;
    const float* x1   = (const float*)d_in[0];
    const float* h0   = (const float*)d_in[1];
    const float* m1   = (const float*)d_in[2];
    const float* d1   = (const float*)d_in[3];
    const float* x_m  = (const float*)d_in[4];
    const float* age  = (const float*)d_in[5];
    const float* gen  = (const float*)d_in[6];
    const int*   ilen = (const int*)d_in[7];
    const int*   txts = (const int*)d_in[8];
    const int*   tlen = (const int*)d_in[9];
    const float* find = (const float*)d_in[10];
    const float* emb  = (const float*)d_in[11];
    const float* Win  = (const float*)d_in[12];
    const float* bin_ = (const float*)d_in[13];
    const float* lng  = (const float*)d_in[14];
    const float* lnb  = (const float*)d_in[15];
    const float* Wq   = (const float*)d_in[16];
    const float* bq   = (const float*)d_in[17];
    const float* Wk   = (const float*)d_in[18];
    const float* bk   = (const float*)d_in[19];
    const float* Wv   = (const float*)d_in[20];
    const float* bv   = (const float*)d_in[21];
    const float* Wo   = (const float*)d_in[22];
    const float* bo   = (const float*)d_in[23];
    const float* ln1g = (const float*)d_in[24];
    const float* ln1b = (const float*)d_in[25];
    const float* Wf1  = (const float*)d_in[26];
    const float* bf1  = (const float*)d_in[27];
    const float* Wf2  = (const float*)d_in[28];
    const float* bf2  = (const float*)d_in[29];
    const float* ln2g = (const float*)d_in[30];
    const float* ln2b = (const float*)d_in[31];
    const float* decw = (const float*)d_in[32];
    const float* decb = (const float*)d_in[33];
    const float* hdW  = (const float*)d_in[34];
    const float* hdb  = (const float*)d_in[35];
    const float* Wih  = (const float*)d_in[36];
    const float* bih  = (const float*)d_in[37];
    const float* Whh  = (const float*)d_in[38];
    const float* bhh  = (const float*)d_in[39];
    const float* lncg = (const float*)d_in[40];
    const float* lncb = (const float*)d_in[41];
    const float* ctw  = (const float*)d_in[42];
    const float* fcW1 = (const float*)d_in[43];
    const float* fcb1 = (const float*)d_in[44];
    const float* bng  = (const float*)d_in[45];
    const float* bnb  = (const float*)d_in[46];
    const float* fcW2 = (const float*)d_in[47];
    const float* fcb2 = (const float*)d_in[48];
    float* out = (float*)d_out;

    // ----- workspace layout -----
    size_t off = 0;
    char* wsb = (char*)d_ws;
    auto alloc = [&](size_t bytes) -> void* {
        void* p = wsb + off;
        off = (off + bytes + 255) & ~(size_t)255;
        return p;
    };
    float*  hdd   = (float*)alloc((size_t)BLr * Hd * 4);
    float*  bufQ  = (float*)alloc((size_t)BLr * Hd * 4);
    float*  bufK  = (float*)alloc((size_t)BLr * Hd * 4);
    float*  bufV  = (float*)alloc((size_t)BLr * Hd * 4);
    float*  bufT  = (float*)alloc((size_t)BLr * FFd * 4);
    __bf16* WinT  = (__bf16*)alloc((size_t)Hd * Hd * 2);
    __bf16* WqT   = (__bf16*)alloc((size_t)NLr * Hd * Hd * 2);
    __bf16* WkT   = (__bf16*)alloc((size_t)NLr * Hd * Hd * 2);
    __bf16* WvT   = (__bf16*)alloc((size_t)NLr * Hd * Hd * 2);
    __bf16* WoT   = (__bf16*)alloc((size_t)NLr * Hd * Hd * 2);
    __bf16* Wf1T  = (__bf16*)alloc((size_t)NLr * FFd * Hd * 2);
    __bf16* Wf2T  = (__bf16*)alloc((size_t)NLr * Hd * FFd * 2);
    __bf16* WihT  = (__bf16*)alloc((size_t)768 * 32 * 2);
    __bf16* WhhT  = (__bf16*)alloc((size_t)768 * 256 * 2);
    __bf16* hdWT  = (__bf16*)alloc((size_t)256 * 32 * 2);
    __bf16* ctwT  = (__bf16*)alloc((size_t)(TSd * Fd) * CPAD * 2);
    __bf16* fc1T  = (__bf16*)alloc((size_t)(TSd * 64) * CPAD * 2);
    float*  Ubuf  = (float*)alloc((size_t)T24d * Bd * 32 * 4);
    float*  Dbuf  = (float*)alloc((size_t)T24d * Bd * 32 * 4);
    float*  gxA   = (float*)alloc((size_t)T24d * Bd * 768 * 4);
    float*  hdcyA = (float*)alloc((size_t)T24d * Bd * Hd * 4);
    float*  ys    = (float*)alloc((size_t)T24d * Bd * Hd * 4);
    float*  hbuf  = (float*)alloc((size_t)Bd * Hd * 4);
    float*  h1buf = (float*)alloc((size_t)Bd * Hd * 4);
    float*  ghbuf = (float*)alloc((size_t)Bd * 768 * 4);
    float*  cpad  = (float*)alloc((size_t)Bd * CPAD * 4);
    float*  pred  = (float*)alloc((size_t)Bd * TSd * Fd * 4);
    float*  z1    = (float*)alloc((size_t)Bd * 768 * 4);
    float*  muv   = (float*)alloc((size_t)768 * 4);
    float*  varv  = (float*)alloc((size_t)768 * 4);

    // ----- weight prep (bf16, transposed, K zero-padded) -----
    transpose_w(stream, Win, WinT, Hd, Hd, Hd);
    for (int l = 0; l < NLr; ++l) {
        transpose_w(stream, Wq + (size_t)l * Hd * Hd, WqT + (size_t)l * Hd * Hd, Hd, Hd, Hd);
        transpose_w(stream, Wk + (size_t)l * Hd * Hd, WkT + (size_t)l * Hd * Hd, Hd, Hd, Hd);
        transpose_w(stream, Wv + (size_t)l * Hd * Hd, WvT + (size_t)l * Hd * Hd, Hd, Hd, Hd);
        transpose_w(stream, Wo + (size_t)l * Hd * Hd, WoT + (size_t)l * Hd * Hd, Hd, Hd, Hd);
        transpose_w(stream, Wf1 + (size_t)l * Hd * FFd, Wf1T + (size_t)l * FFd * Hd, Hd, FFd, Hd);
        transpose_w(stream, Wf2 + (size_t)l * FFd * Hd, Wf2T + (size_t)l * Hd * FFd, FFd, Hd, FFd);
    }
    transpose_w(stream, Wih, WihT, 32, 768, 32);
    transpose_w(stream, Whh, WhhT, Hd, 768, Hd);
    transpose_w(stream, hdW, hdWT, Fd, Hd, 32);
    {
        long tot = (long)TSd * Fd * CPAD;
        transpose_tcd_bf16_kernel<<<(int)((tot + 255) / 256), 256, 0, stream>>>(
            ctw, ctwT, TSd, CDIMd, Fd, CPAD);
        tot = (long)TSd * 64 * CPAD;
        transpose_tcd_bf16_kernel<<<(int)((tot + 255) / 256), 256, 0, stream>>>(
            fcW1, fc1T, TSd, CDIMd, 64, CPAD);
    }

    // ----- transformer input: emb gather -> @Win -> LN + PE -----
    gather_emb_kernel<<<(BLr * Hd + 255) / 256, 256, 0, stream>>>(txts, emb, bufT);
    gemm(stream, bufT, Hd, WinT, Hd, bin_, bufQ, Hd, BLr, Hd, ACT_NONE);
    ln256_kernel<<<(BLr + 7) / 8, 256, 0, stream>>>(bufQ, nullptr, lng, lnb, hdd, BLr, 2);

    // ----- encoder layers -----
    for (int l = 0; l < NLr; ++l) {
        size_t wo = (size_t)l * Hd * Hd;
        gemm(stream, hdd, Hd, WqT + wo, Hd, bq + l * Hd, bufQ, Hd, BLr, Hd, ACT_NONE);
        gemm(stream, hdd, Hd, WkT + wo, Hd, bk + l * Hd, bufK, Hd, BLr, Hd, ACT_NONE);
        gemm(stream, hdd, Hd, WvT + wo, Hd, bv + l * Hd, bufV, Hd, BLr, Hd, ACT_NONE);
        attn_kernel<<<Bd * NHh, 256, 0, stream>>>(bufQ, bufK, bufV, tlen, bufT);
        gemm(stream, bufT, Hd, WoT + wo, Hd, bo + l * Hd, bufQ, Hd, BLr, Hd, ACT_NONE);
        ln256_kernel<<<(BLr + 7) / 8, 256, 0, stream>>>(hdd, bufQ, ln1g + l * Hd, ln1b + l * Hd,
                                                        hdd, BLr, 1);
        gemm(stream, hdd, Hd, Wf1T + (size_t)l * FFd * Hd, Hd, bf1 + l * FFd, bufT, FFd,
             BLr, FFd, ACT_RELU);
        gemm(stream, bufT, FFd, Wf2T + (size_t)l * Hd * FFd, FFd, bf2 + l * Hd, bufQ, Hd,
             BLr, Hd, ACT_NONE);
        ln256_kernel<<<(BLr + 7) / 8, 256, 0, stream>>>(hdd, bufQ, ln2g + l * Hd, ln2b + l * Hd,
                                                        hdd, BLr, 1);
    }

    // ----- GRU-D branch -----
    xprep_kernel<<<(T24d * Bd * Fd + 255) / 256, 256, 0, stream>>>(x1, m1, d1, x_m, decw, decb,
                                                                   Ubuf, Dbuf);
    gemm(stream, Ubuf, 32, WihT, 32, bih, gxA, 768, T24d * Bd, 768, ACT_NONE);
    gemm(stream, Dbuf, 32, hdWT, 32, hdb, hdcyA, Hd, T24d * Bd, Hd, ACT_EXPNEGRELU);
    copyf_kernel<<<(Bd * Hd + 255) / 256, 256, 0, stream>>>(h0, hbuf, Bd * Hd);
    for (int t = 0; t < T24d; ++t) {
        gru_decay_kernel<<<(Bd * Hd + 255) / 256, 256, 0, stream>>>(
            hdcyA + (size_t)t * Bd * Hd, hbuf, h1buf);
        gemm(stream, h1buf, Hd, WhhT, Hd, bhh, ghbuf, 768, Bd, 768, ACT_NONE);
        gru_gate_kernel<<<(Bd * Hd + 255) / 256, 256, 0, stream>>>(
            gxA + (size_t)t * Bd * 768, ghbuf, h1buf, hbuf, ys + (size_t)t * Bd * Hd);
    }

    // ----- heads -----
    build_ct_kernel<<<Bd, 256, 0, stream>>>(ys, hdd, age, gen, ilen, lncg, lncb, cpad);
    gemm(stream, cpad, CPAD, ctwT, CPAD, nullptr, pred, TSd * Fd, Bd, TSd * Fd, ACT_NONE);
    loss_kernel<<<1, 256, 0, stream>>>(pred, x1, find, out + TSd * Bd);
    gemm(stream, cpad, CPAD, fc1T, CPAD, fcb1, z1, TSd * 64, Bd, TSd * 64, ACT_NONE);
    bnstat_kernel<<<3, 256, 0, stream>>>(z1, muv, varv);
    head_kernel<<<(TSd * Bd + 255) / 256, 256, 0, stream>>>(z1, muv, varv, bng, bnb, fcW2,
                                                            fcb2, out);
}
